// RWKVLayer_15298673508623
// MI455X (gfx1250) — compile-verified, matrix operand order checked by reading, not run on hardware
//
#include <hip/hip_runtime.h>
#include <hip/hip_bf16.h>
#include <cstdint>
#include <cstddef>

// ---------------- model constants ----------------
#define D_MODEL   2048
#define NHEADS    32
#define HEADSZ    64
#define SEQ_T     2048
#define LORA_DIM  64
#define GLORA_DIM 128
#define FFN_DIM   (4 * D_MODEL)
#define EPS_LN    1e-5f
#define EPS_GN    64e-5f

typedef __attribute__((ext_vector_type(16))) __bf16 v16bf;
typedef __attribute__((ext_vector_type(8)))  __bf16 v8bf;
typedef __attribute__((ext_vector_type(8)))  float  v8f;

__device__ __forceinline__ __bf16 to_bf16(float f) { return (__bf16)f; }

// ---------------------------------------------------------------------------
// Weight transpose + fp32 -> bf16 convert.  W is [K x N] row-major fp32,
// output Wt is [N x K] row-major bf16 (so GEMM B-operand loads are per-lane
// contiguous along K).
// ---------------------------------------------------------------------------
__global__ __launch_bounds__(256)
void transpose_to_bf16(const float* __restrict__ W, __bf16* __restrict__ Wt,
                       int K, int N)
{
    __shared__ float tile[32][33];
    const int nb = blockIdx.x * 32;
    const int kb = blockIdx.y * 32;
    const int tx = threadIdx.x;   // 0..31
    const int ty = threadIdx.y;   // 0..7
#pragma unroll
    for (int j = 0; j < 4; ++j)
        tile[ty + 8 * j][tx] = W[(size_t)(kb + ty + 8 * j) * N + nb + tx];
    __syncthreads();
#pragma unroll
    for (int j = 0; j < 4; ++j)
        Wt[(size_t)(nb + ty + 8 * j) * K + kb + tx] = to_bf16(tile[tx][ty + 8 * j]);
}

// ---------------------------------------------------------------------------
// Per-token LayerNorm over D_MODEL channels (population variance, EPS_LN).
// ---------------------------------------------------------------------------
__global__ __launch_bounds__(256)
void layernorm_kernel(const float* __restrict__ x, const float* __restrict__ w,
                      const float* __restrict__ b, float* __restrict__ out)
{
    const int t = blockIdx.x;
    const float* xr = x + (size_t)t * D_MODEL;
    float* orow = out + (size_t)t * D_MODEL;
    float s = 0.f, s2 = 0.f;
    for (int i = threadIdx.x; i < D_MODEL; i += 256) {
        float v = xr[i]; s += v; s2 += v * v;
    }
    __shared__ float rs[256], rs2[256];
    rs[threadIdx.x] = s; rs2[threadIdx.x] = s2;
    __syncthreads();
    for (int off = 128; off > 0; off >>= 1) {
        if (threadIdx.x < off) {
            rs[threadIdx.x]  += rs[threadIdx.x + off];
            rs2[threadIdx.x] += rs2[threadIdx.x + off];
        }
        __syncthreads();
    }
    const float mean = rs[0] * (1.f / D_MODEL);
    const float var  = rs2[0] * (1.f / D_MODEL) - mean * mean;
    const float inv  = rsqrtf(var + EPS_LN);
    for (int i = threadIdx.x; i < D_MODEL; i += 256)
        orow[i] = (xr[i] - mean) * inv * w[i] + b[i];
}

// ---------------------------------------------------------------------------
// out[t,d] = bf16( x[t,d] + (x[t-1,d] - x[t,d]) * coeff[d] ),  x[-1] = 0.
// ---------------------------------------------------------------------------
__global__ __launch_bounds__(256)
void shift_mix_kernel(const float* __restrict__ x, const float* __restrict__ coeff,
                      __bf16* __restrict__ out)
{
    const size_t idx = (size_t)blockIdx.x * 256 + threadIdx.x;
    const int d = (int)(idx & (D_MODEL - 1));
    const float cur  = x[idx];
    const float prev = (idx >= D_MODEL) ? x[idx - D_MODEL] : 0.f;
    out[idx] = to_bf16(cur + (prev - cur) * coeff[d]);
}

// ---------------------------------------------------------------------------
// Elementwise fp32 -> bf16 with optional activation.
// mode: 0 = identity, 1 = tanh, 2 = sigmoid, 3 = relu^2
// ---------------------------------------------------------------------------
__global__ __launch_bounds__(256)
void unary_to_bf16_kernel(const float* __restrict__ in, __bf16* __restrict__ out,
                          int n, int mode)
{
    const int idx = blockIdx.x * 256 + threadIdx.x;
    if (idx >= n) return;
    float v = in[idx];
    if (mode == 1)      v = tanhf(v);
    else if (mode == 2) v = 1.f / (1.f + expf(-v));
    else if (mode == 3) { v = fmaxf(v, 0.f); v = v * v; }
    out[idx] = to_bf16(v);
}

// ---------------------------------------------------------------------------
// Main bf16 WMMA GEMM (block tile 128x128, K-step 32, LDS double-buffered):
//   C[M x N] (fp32) = A[M x K] (bf16 row-major) @ Bt[N x K]^T (bf16).
// Optional fused residual: C = addend + A@B.
// 256 threads = 8 waves; wave (mw = wv>>1, nw = wv&1) computes a 32x64 strip
// as 2x4 v_wmma_f32_16x16x32_bf16 tiles; all 6 fragments are loaded before
// the 8 WMMAs so they can issue back-to-back into the matrix pipe.
// Fragment layout per CDNA5 ISA 7.12.2: lane l holds row/col (l&15), l[4]
// selects the K half-groups {8h..8h+7, 16+8h..16+8h+7}.
// ---------------------------------------------------------------------------
__global__ __launch_bounds__(256)
void gemm_bf16_wmma_128(const __bf16* __restrict__ A, const __bf16* __restrict__ Bt,
                        const float* __restrict__ addend, float* __restrict__ C,
                        int N, int K)
{
    __shared__ alignas(32) __bf16 aT[2][128][32];   // 2 x 8 KB
    __shared__ alignas(32) __bf16 bT[2][128][32];   // 2 x 8 KB

    const int tid  = threadIdx.x;       // 0..255
    const int wv   = tid >> 5;          // 0..7
    const int lane = tid & 31;
    const int l16  = lane & 15;
    const int hi   = lane >> 4;
    const int mw   = wv >> 1;           // 0..3 : 32-row strip
    const int nw   = wv & 1;            // 0..1 : 64-col strip
    const int m0   = blockIdx.y * 128;
    const int n0   = blockIdx.x * 128;
    const int ldr  = tid >> 1;          // 0..127 staging row
    const int ldc  = (tid & 1) * 16;    // 16-element chunk in the K tile

    const __bf16* Aload = &A [(size_t)(m0 + ldr) * K + ldc];
    const __bf16* Bload = &Bt[(size_t)(n0 + ldr) * K + ldc];

    v8f acc[2][4] = {};

    // preload K-tile 0 into buffer 0
    *(v16bf*)&aT[0][ldr][ldc] = *(const v16bf*)&Aload[0];
    *(v16bf*)&bT[0][ldr][ldc] = *(const v16bf*)&Bload[0];
    __syncthreads();

    for (int kb = 0; kb < K; kb += 32) {
        const int cur = (kb >> 5) & 1;
        if (kb + 32 < K) {   // stage next tile into the other buffer
            *(v16bf*)&aT[cur ^ 1][ldr][ldc] = *(const v16bf*)&Aload[kb + 32];
            *(v16bf*)&bT[cur ^ 1][ldr][ldc] = *(const v16bf*)&Bload[kb + 32];
            if (kb + 64 < K) {   // hint the tile after that into cache
                __builtin_prefetch(&Aload[kb + 64], 0, 1);
                __builtin_prefetch(&Bload[kb + 64], 0, 1);
            }
        }

        union Frag { v16bf v; v8bf h[2]; } af[2], bfr[4];
#pragma unroll
        for (int fa = 0; fa < 2; ++fa) {
            const int r = mw * 32 + fa * 16 + l16;
            af[fa].h[0] = *(const v8bf*)&aT[cur][r][8 * hi];
            af[fa].h[1] = *(const v8bf*)&aT[cur][r][16 + 8 * hi];
        }
#pragma unroll
        for (int fb = 0; fb < 4; ++fb) {
            const int r = nw * 64 + fb * 16 + l16;
            bfr[fb].h[0] = *(const v8bf*)&bT[cur][r][8 * hi];
            bfr[fb].h[1] = *(const v8bf*)&bT[cur][r][16 + 8 * hi];
        }
#pragma unroll
        for (int fa = 0; fa < 2; ++fa)
#pragma unroll
            for (int fb = 0; fb < 4; ++fb)
                acc[fa][fb] = __builtin_amdgcn_wmma_f32_16x16x32_bf16(
                    false, af[fa].v, false, bfr[fb].v, (short)0, acc[fa][fb],
                    false, false);
        __syncthreads();
    }

#pragma unroll
    for (int fa = 0; fa < 2; ++fa)
#pragma unroll
        for (int fb = 0; fb < 4; ++fb)
#pragma unroll
            for (int i = 0; i < 8; ++i) {
                const size_t row = (size_t)m0 + mw * 32 + fa * 16 + 8 * hi + i;
                const size_t col = (size_t)n0 + nw * 64 + fb * 16 + l16;
                const size_t off = row * (size_t)N + col;
                float val = acc[fa][fb][i];
                if (addend) val += addend[off];
                C[off] = val;
            }
}

// ---------------------------------------------------------------------------
// Narrow-N GEMM (block tile 64x64) for the N=64 LoRA projections.
// ---------------------------------------------------------------------------
__global__ __launch_bounds__(128)
void gemm_bf16_wmma_64(const __bf16* __restrict__ A, const __bf16* __restrict__ Bt,
                       const float* __restrict__ addend, float* __restrict__ C,
                       int N, int K)
{
    __shared__ alignas(32) __bf16 aT[64][32];
    __shared__ alignas(32) __bf16 bT[64][32];

    const int tid  = threadIdx.x;
    const int wv   = tid >> 5;        // wave 0..3
    const int lane = tid & 31;
    const int l16  = lane & 15;
    const int hi   = lane >> 4;
    const int m0   = blockIdx.y * 64;
    const int n0   = blockIdx.x * 64;
    const int ldr  = tid >> 1;
    const int ldc  = (tid & 1) * 16;

    v8f acc[4] = {};

    for (int kb = 0; kb < K; kb += 32) {
        *(v16bf*)&aT[ldr][ldc] = *(const v16bf*)&A [(size_t)(m0 + ldr) * K + kb + ldc];
        *(v16bf*)&bT[ldr][ldc] = *(const v16bf*)&Bt[(size_t)(n0 + ldr) * K + kb + ldc];
        __syncthreads();

        union Frag { v16bf v; v8bf h[2]; } af, bfr[4];
        af.h[0] = *(const v8bf*)&aT[16 * wv + l16][8 * hi];
        af.h[1] = *(const v8bf*)&aT[16 * wv + l16][16 + 8 * hi];
#pragma unroll
        for (int j = 0; j < 4; ++j) {
            bfr[j].h[0] = *(const v8bf*)&bT[16 * j + l16][8 * hi];
            bfr[j].h[1] = *(const v8bf*)&bT[16 * j + l16][16 + 8 * hi];
        }
#pragma unroll
        for (int j = 0; j < 4; ++j)
            acc[j] = __builtin_amdgcn_wmma_f32_16x16x32_bf16(
                false, af.v, false, bfr[j].v, (short)0, acc[j], false, false);
        __syncthreads();
    }

#pragma unroll
    for (int j = 0; j < 4; ++j)
#pragma unroll
        for (int i = 0; i < 8; ++i) {
            const size_t row = (size_t)m0 + 16 * wv + 8 * hi + i;
            const size_t col = (size_t)n0 + 16 * j + l16;
            const size_t off = row * (size_t)N + col;
            float val = acc[j][i];
            if (addend) val += addend[off];
            C[off] = val;
        }
}

// ---------------------------------------------------------------------------
// Pre-scan elementwise, one block of 64 threads per (t, h).
// In-place: kf (raw k -> final k), vf (raw v -> blended v),
//           wlora (-> wexp), alora (-> a).  kkout written fresh.
// ---------------------------------------------------------------------------
__global__ __launch_bounds__(64)
void prescan_kernel(float* __restrict__ kf, float* __restrict__ vf,
                    float* __restrict__ wlora, float* __restrict__ alora,
                    const float* __restrict__ vlora, const float* __restrict__ v_first,
                    const float* __restrict__ k_k, const float* __restrict__ k_a,
                    const float* __restrict__ w0, const float* __restrict__ a0,
                    const float* __restrict__ v0, float* __restrict__ kkout)
{
    const int th = blockIdx.x;          // t*NHEADS + h
    const int n  = threadIdx.x;
    const int h  = th & (NHEADS - 1);
    const int d  = h * HEADSZ + n;
    const size_t idx = (size_t)th * HEADSZ + n;   // == t*D + d

    const float kraw = kf[idx];
    const float kkv  = kraw * k_k[d];
    __shared__ float red[64];
    red[n] = kkv * kkv;
    __syncthreads();
    for (int off = 32; off > 0; off >>= 1) {
        if (n < off) red[n] += red[n + off];
        __syncthreads();
    }
    const float nrm = fmaxf(sqrtf(red[0]), 1e-12f);
    kkout[idx] = kkv / nrm;

    const float wl   = w0[d] + wlora[idx];
    const float wneg = -log1pf(expf(-wl)) - 0.5f;   // -softplus(-wl) - 0.5
    wlora[idx] = expf(-expf(wneg));                 // wexp

    const float av = 1.f / (1.f + expf(-(a0[d] + alora[idx])));
    alora[idx] = av;

    const float vraw = vf[idx];
    const float vs   = 1.f / (1.f + expf(-(v0[d] + vlora[idx])));
    vf[idx] = vraw + (v_first[idx] - vraw) * vs;

    kf[idx] = kraw * (1.f + (av - 1.f) * k_a[d]);
}

// ---------------------------------------------------------------------------
// Sequential WKV-7 scan.  One block per head; 256 threads each hold 16
// elements of the 64x64 fp32 state S[v][k] in registers:
//   thread (v = tid&63, kg = tid>>6) owns S[v][16*kg .. 16*kg+15].
// ---------------------------------------------------------------------------
__global__ __launch_bounds__(256)
void wkv_scan_kernel(const float* __restrict__ r, const float* __restrict__ wexp,
                     const float* __restrict__ k, const float* __restrict__ v,
                     const float* __restrict__ kk, const float* __restrict__ a,
                     float* __restrict__ y)
{
    const int h    = blockIdx.x;      // 0..NHEADS-1
    const int tid  = threadIdx.x;
    const int vrow = tid & 63;
    const int kg   = tid >> 6;        // 0..3
    const int k0   = kg * 16;

    float S[16];
#pragma unroll
    for (int j = 0; j < 16; ++j) S[j] = 0.f;

    __shared__ float rs[64], we[64], ks[64], vs[64], kks[64], as[64];
    __shared__ float part[64][4];

    for (int t = 0; t < SEQ_T; ++t) {
        const size_t base = ((size_t)t * NHEADS + h) * HEADSZ;
        if (tid < 64) {
            rs[tid]  = r[base + tid];   we[tid]  = wexp[base + tid];
            ks[tid]  = k[base + tid];   vs[tid]  = v[base + tid];
            kks[tid] = kk[base + tid];  as[tid]  = a[base + tid];
        }
        __syncthreads();

        float p = 0.f;
#pragma unroll
        for (int j = 0; j < 16; ++j) p += S[j] * kks[k0 + j];
        part[vrow][kg] = p;
        __syncthreads();
        const float sa = part[vrow][0] + part[vrow][1] + part[vrow][2] + part[vrow][3];

        const float vv = vs[vrow];
        float p2 = 0.f;
#pragma unroll
        for (int j = 0; j < 16; ++j) {
            const int kc = k0 + j;
            const float Sn = S[j] * we[kc] - sa * (kks[kc] * as[kc]) + vv * ks[kc];
            S[j] = Sn;
            p2 += Sn * rs[kc];
        }
        __syncthreads();               // all sa reads of part[] done
        part[vrow][kg] = p2;
        __syncthreads();
        if (kg == 0)
            y[base + vrow] = part[vrow][0] + part[vrow][1] + part[vrow][2] + part[vrow][3];
        __syncthreads();               // protect shared vectors for next t
    }
}

// ---------------------------------------------------------------------------
// Post-scan: per-(t,h) GroupNorm over 64 channels (EPS_GN) + r.k.r_k bonus,
// then multiply by gate g and emit bf16 for the Wo GEMM.
// ---------------------------------------------------------------------------
__global__ __launch_bounds__(64)
void postscan_kernel(const float* __restrict__ y, const float* __restrict__ r,
                     const float* __restrict__ kfin, const float* __restrict__ vfin,
                     const float* __restrict__ g, const float* __restrict__ r_k,
                     const float* __restrict__ lnx_w, const float* __restrict__ lnx_b,
                     __bf16* __restrict__ out)
{
    const int th = blockIdx.x;
    const int n  = threadIdx.x;
    const int h  = th & (NHEADS - 1);
    const int d  = h * HEADSZ + n;
    const size_t idx = (size_t)th * HEADSZ + n;

    const float yv = y[idx];
    __shared__ float rs[64], rs2[64], rkr[64];
    rs[n]  = yv;
    rs2[n] = yv * yv;
    rkr[n] = r[idx] * kfin[idx] * r_k[d];
    __syncthreads();
    for (int off = 32; off > 0; off >>= 1) {
        if (n < off) {
            rs[n]  += rs[n + off];
            rs2[n] += rs2[n + off];
            rkr[n] += rkr[n + off];
        }
        __syncthreads();
    }
    const float mu  = rs[0] * (1.f / HEADSZ);
    const float var = rs2[0] * (1.f / HEADSZ) - mu * mu;
    float yn = (yv - mu) * rsqrtf(var + EPS_GN) * lnx_w[d] + lnx_b[d];
    yn += rkr[0] * vfin[idx];
    out[idx] = to_bf16(yn * g[idx]);
}

// ---------------------------------------------------------------------------
// Host-side orchestration
// ---------------------------------------------------------------------------
extern "C" void kernel_launch(void* const* d_in, const int* in_sizes, int n_in,
                              void* d_out, int out_size, void* d_ws, size_t ws_size,
                              hipStream_t stream)
{
    (void)in_sizes; (void)n_in; (void)out_size; (void)ws_size;

    const float* x_in   = (const float*)d_in[0];
    const float* vfirst = (const float*)d_in[1];
    const float* x_r = (const float*)d_in[2];
    const float* x_w = (const float*)d_in[3];
    const float* x_k = (const float*)d_in[4];
    const float* x_v = (const float*)d_in[5];
    const float* x_a = (const float*)d_in[6];
    const float* x_g = (const float*)d_in[7];
    const float* k_k = (const float*)d_in[8];
    const float* k_a = (const float*)d_in[9];
    const float* w0  = (const float*)d_in[10];
    const float* w1  = (const float*)d_in[11];
    const float* w2  = (const float*)d_in[12];
    const float* a0  = (const float*)d_in[13];
    const float* a1  = (const float*)d_in[14];
    const float* a2  = (const float*)d_in[15];
    const float* v0  = (const float*)d_in[16];
    const float* v1  = (const float*)d_in[17];
    const float* v2  = (const float*)d_in[18];
    const float* g1  = (const float*)d_in[19];
    const float* g2  = (const float*)d_in[20];
    const float* r_k = (const float*)d_in[21];
    const float* Wr  = (const float*)d_in[22];
    const float* Wk  = (const float*)d_in[23];
    const float* Wv  = (const float*)d_in[24];
    const float* Wo  = (const float*)d_in[25];
    const float* lnx_w = (const float*)d_in[26];
    const float* lnx_b = (const float*)d_in[27];
    // d_in[28], d_in[29]: ln0 (unused, LAYER_IDX==1)
    const float* ln1_w = (const float*)d_in[30];
    const float* ln1_b = (const float*)d_in[31];
    const float* ln2_w = (const float*)d_in[32];
    const float* ln2_b = (const float*)d_in[33];
    const float* cm_x_k = (const float*)d_in[34];
    const float* cmWk = (const float*)d_in[35];
    const float* cmWv = (const float*)d_in[36];
    // d_in[37]: state_update flag (unused by full-sequence reference)

    const size_t TD = (size_t)SEQ_T * D_MODEL;

    char* wp = (char*)d_ws;
    auto alloc = [&](size_t bytes) -> char* {
        char* r = wp;
        wp += (bytes + 255) & ~(size_t)255;
        return r;
    };

    // bf16 transposed weights
    __bf16* Wr_t   = (__bf16*)alloc((size_t)D_MODEL * D_MODEL * 2);
    __bf16* Wk_t   = (__bf16*)alloc((size_t)D_MODEL * D_MODEL * 2);
    __bf16* Wv_t   = (__bf16*)alloc((size_t)D_MODEL * D_MODEL * 2);
    __bf16* Wo_t   = (__bf16*)alloc((size_t)D_MODEL * D_MODEL * 2);
    __bf16* w1_t   = (__bf16*)alloc((size_t)LORA_DIM * D_MODEL * 2);
    __bf16* w2_t   = (__bf16*)alloc((size_t)D_MODEL * LORA_DIM * 2);
    __bf16* a1_t   = (__bf16*)alloc((size_t)LORA_DIM * D_MODEL * 2);
    __bf16* a2_t   = (__bf16*)alloc((size_t)D_MODEL * LORA_DIM * 2);
    __bf16* v1_t   = (__bf16*)alloc((size_t)LORA_DIM * D_MODEL * 2);
    __bf16* v2_t   = (__bf16*)alloc((size_t)D_MODEL * LORA_DIM * 2);
    __bf16* g1_t   = (__bf16*)alloc((size_t)GLORA_DIM * D_MODEL * 2);
    __bf16* g2_t   = (__bf16*)alloc((size_t)D_MODEL * GLORA_DIM * 2);
    __bf16* cmWk_t = (__bf16*)alloc((size_t)FFN_DIM * D_MODEL * 2);
    __bf16* cmWv_t = (__bf16*)alloc((size_t)D_MODEL * FFN_DIM * 2);

    // fp32 activations
    float* xln   = (float*)alloc(TD * 4);   // ln1(x); later x after Wo residual
    float* rF    = (float*)alloc(TD * 4);
    float* kF    = (float*)alloc(TD * 4);   // raw k -> final k (in place)
    float* vF    = (float*)alloc(TD * 4);   // raw v -> blended v (in place)
    float* wlora = (float*)alloc(TD * 4);   // -> wexp (in place)
    float* alora = (float*)alloc(TD * 4);   // -> a (in place)
    float* vlora = (float*)alloc(TD * 4);   // -> y (scan out) -> xl (ln2 out)
    float* gF    = (float*)alloc(TD * 4);
    float* kkT   = (float*)alloc(TD * 4);
    float* wm    = (float*)alloc((size_t)SEQ_T * LORA_DIM * 4);
    float* am    = (float*)alloc((size_t)SEQ_T * LORA_DIM * 4);
    float* vm    = (float*)alloc((size_t)SEQ_T * LORA_DIM * 4);
    float* gm    = (float*)alloc((size_t)SEQ_T * GLORA_DIM * 4);
    float* hid   = (float*)alloc((size_t)SEQ_T * FFN_DIM * 4);

    // bf16 activations
    __bf16* xb   = (__bf16*)alloc(TD * 2);                         // mixes/yng/ck
    __bf16* midb = (__bf16*)alloc((size_t)SEQ_T * GLORA_DIM * 2);  // lora mids
    __bf16* hidb = (__bf16*)alloc((size_t)SEQ_T * FFN_DIM * 2);

    const dim3 trb(32, 8);
    auto tr = [&](const float* W, __bf16* Wt, int K, int N) {
        transpose_to_bf16<<<dim3(N / 32, K / 32), trb, 0, stream>>>(W, Wt, K, N);
    };
    auto gemm = [&](const __bf16* A, const __bf16* Bt, const float* add,
                    float* C, int M, int N, int K) {
        if (N % 128 == 0)
            gemm_bf16_wmma_128<<<dim3(N / 128, M / 128), 256, 0, stream>>>(
                A, Bt, add, C, N, K);
        else
            gemm_bf16_wmma_64<<<dim3(N / 64, M / 64), 128, 0, stream>>>(
                A, Bt, add, C, N, K);
    };
    auto unary = [&](const float* in, __bf16* out, int n, int mode) {
        unary_to_bf16_kernel<<<(n + 255) / 256, 256, 0, stream>>>(in, out, n, mode);
    };
    const int mixBlocks = (int)(TD / 256);
    auto mix = [&](const float* src, const float* coeff) {
        shift_mix_kernel<<<mixBlocks, 256, 0, stream>>>(src, coeff, xb);
    };

    // 1) weight conversion/transpose (reused across all T=2048 rows)
    tr(Wr, Wr_t, D_MODEL, D_MODEL);
    tr(Wk, Wk_t, D_MODEL, D_MODEL);
    tr(Wv, Wv_t, D_MODEL, D_MODEL);
    tr(Wo, Wo_t, D_MODEL, D_MODEL);
    tr(w1, w1_t, D_MODEL, LORA_DIM);
    tr(w2, w2_t, LORA_DIM, D_MODEL);
    tr(a1, a1_t, D_MODEL, LORA_DIM);
    tr(a2, a2_t, LORA_DIM, D_MODEL);
    tr(v1, v1_t, D_MODEL, LORA_DIM);
    tr(v2, v2_t, LORA_DIM, D_MODEL);
    tr(g1, g1_t, D_MODEL, GLORA_DIM);
    tr(g2, g2_t, GLORA_DIM, D_MODEL);

    // 2) ln1
    layernorm_kernel<<<SEQ_T, 256, 0, stream>>>(x_in, ln1_w, ln1_b, xln);

    // 3) r / w / k branches
    mix(xln, x_r);
    gemm(xb, Wr_t, nullptr, rF, SEQ_T, D_MODEL, D_MODEL);
    mix(xln, x_w);
    gemm(xb, w1_t, nullptr, wm, SEQ_T, LORA_DIM, D_MODEL);
    unary(wm, midb, SEQ_T * LORA_DIM, 1);                         // tanh
    gemm(midb, w2_t, nullptr, wlora, SEQ_T, D_MODEL, LORA_DIM);
    mix(xln, x_k);
    gemm(xb, Wk_t, nullptr, kF, SEQ_T, D_MODEL, D_MODEL);

    // 4) v branch (shared xv mix feeds Wv and v-lora)
    mix(xln, x_v);
    gemm(xb, Wv_t, nullptr, vF, SEQ_T, D_MODEL, D_MODEL);
    gemm(xb, v1_t, nullptr, vm, SEQ_T, LORA_DIM, D_MODEL);
    unary(vm, midb, SEQ_T * LORA_DIM, 0);
    gemm(midb, v2_t, nullptr, vlora, SEQ_T, D_MODEL, LORA_DIM);

    // 5) a branch
    mix(xln, x_a);
    gemm(xb, a1_t, nullptr, am, SEQ_T, LORA_DIM, D_MODEL);
    unary(am, midb, SEQ_T * LORA_DIM, 0);
    gemm(midb, a2_t, nullptr, alora, SEQ_T, D_MODEL, LORA_DIM);

    // 6) g branch
    mix(xln, x_g);
    gemm(xb, g1_t, nullptr, gm, SEQ_T, GLORA_DIM, D_MODEL);
    unary(gm, midb, SEQ_T * GLORA_DIM, 2);                        // sigmoid
    gemm(midb, g2_t, nullptr, gF, SEQ_T, D_MODEL, GLORA_DIM);

    // 7) pre-scan elementwise (kk normalize, wexp, a, v blend, final k)
    prescan_kernel<<<SEQ_T * NHEADS, 64, 0, stream>>>(
        kF, vF, wlora, alora, vlora, vfirst, k_k, k_a, w0, a0, v0, kkT);

    // 8) sequential WKV scan -> y (reusing vlora buffer)
    float* yBuf = vlora;
    wkv_scan_kernel<<<NHEADS, 256, 0, stream>>>(rF, wlora, kF, vF, kkT, alora, yBuf);

    // 9) GroupNorm + bonus + gate -> bf16 (reusing xb)
    postscan_kernel<<<SEQ_T * NHEADS, 64, 0, stream>>>(
        yBuf, rF, kF, vF, gF, r_k, lnx_w, lnx_b, xb);

    // 10) x = ln1(x) + (yn*g) @ Wo   (fused residual, in place on xln)
    gemm(xb, Wo_t, xln, xln, SEQ_T, D_MODEL, D_MODEL);

    // 11) ChannelMix
    float* xl = yBuf;   // reuse y buffer for ln2 output
    layernorm_kernel<<<SEQ_T, 256, 0, stream>>>(xln, ln2_w, ln2_b, xl);
    mix(xl, cm_x_k);    // ck -> xb (bf16)
    gemm(xb, cmWk_t, nullptr, hid, SEQ_T, FFN_DIM, D_MODEL);
    unary(hid, hidb, SEQ_T * FFN_DIM, 3);                         // relu^2
    // final: d_out[0:T*D] = x + relu(ck@Wk)^2 @ Wv   (fused residual)
    gemm(hidb, cmWv_t, xln, (float*)d_out, SEQ_T, D_MODEL, FFN_DIM);

    // 12) second output: v_first passes through unchanged (LAYER_IDX != 0)
    hipMemcpyAsync((float*)d_out + TD, vfirst, TD * sizeof(float),
                   hipMemcpyDeviceToDevice, stream);
}
